// Head_40587440947229
// MI455X (gfx1250) — matrix operand order repaired and checked by reading, not compiled, not run on hardware
//
#include <hip/hip_runtime.h>
#include <hip/hip_bf16.h>

// ---------------------------------------------------------------------------
// Causal single-head attention (B=4, T=4096, E=1024, H=64) for gfx1250.
// Compute-bound (~47 GFLOP vs ~100MB traffic) -> all matmuls go through
// v_wmma_f32_16x16x32_f16 with f32 accumulation.
// Pipeline:
//   1) cvt_x_kernel:  x f32 -> xh f16 (one-shot, vectorized)
//   2) cvt_w_kernel:  Wq/Wk/Wv f32 -> Wt f16 transposed [3][HD][E] so that
//      projection B-fragments are two contiguous 16B loads
//   3) qkv_proj_kernel: Q,K f16 row-major; V f16 transposed [B][HD][T] so
//      that PV B-fragments are two contiguous 16B loads
//   4) flash_attn_kernel: online-softmax streaming attention, P staged
//      through LDS to re-layout C-fragment -> A-fragment
// ---------------------------------------------------------------------------

typedef __attribute__((ext_vector_type(16))) _Float16 v16h;
typedef __attribute__((ext_vector_type(8)))  _Float16 v8h;
typedef __attribute__((ext_vector_type(8)))  float    v8f;

namespace {
constexpr int BB = 4;      // batch
constexpr int T  = 4096;   // sequence
constexpr int E  = 1024;   // embed
constexpr int HD = 64;     // head dim
constexpr int BT = BB * T; // 16384 total rows

__device__ inline v16h cat8(v8h lo, v8h hi) {
  return __builtin_shufflevector(lo, hi, 0,1,2,3,4,5,6,7,8,9,10,11,12,13,14,15);
}

// Deterministic per-(b,q,k) dropout decision, keep-prob 0.9.
// (Reference uses JAX threefry key 42 — not bit-reproducible; deterministic
// stand-in with identical statistics.)
__device__ inline bool keep_drop(int b, int q, int k) {
  unsigned h = ((unsigned)(b * T + q) * 0x01000193u) ^ (unsigned)k;
  h *= 0x9E3779B1u;
  h ^= h >> 15;
  h *= 0x85EBCA77u;
  h ^= h >> 13;
  return (h >> 8) < 15099494u;   // 0.9 * 2^24
}
} // namespace

// ---------------------------------------------------------------------------
// Prep 1: x f32 -> f16, 8 elements per thread, vector in / vector out.
// ---------------------------------------------------------------------------
__global__ __launch_bounds__(256) void cvt_x_kernel(
    const float* __restrict__ x, _Float16* __restrict__ xh)
{
  const size_t i = ((size_t)blockIdx.x * blockDim.x + threadIdx.x) * 8;
  v8f v = *(const v8f*)(x + i);
  v8h o;
  #pragma unroll
  for (int j = 0; j < 8; ++j) o[j] = (_Float16)v[j];
  *(v8h*)(xh + i) = o;
}

// ---------------------------------------------------------------------------
// Prep 2: W[E][HD] f32 -> Wt[3][HD][E] f16 (transposed). 196K elements total.
// ---------------------------------------------------------------------------
__global__ __launch_bounds__(256) void cvt_w_kernel(
    const float* __restrict__ Wq, const float* __restrict__ Wk,
    const float* __restrict__ Wv, _Float16* __restrict__ Wt)
{
  const int i = blockIdx.x * blockDim.x + threadIdx.x;   // 0 .. 3*E*HD-1
  const int mat = i / (E * HD);
  const int rem = i - mat * E * HD;
  const int k = rem / HD;
  const int h = rem - k * HD;
  const float* __restrict__ W = (mat == 0) ? Wq : ((mat == 1) ? Wk : Wv);
  Wt[(size_t)mat * HD * E + (size_t)h * E + k] = (_Float16)W[rem];
}

// ---------------------------------------------------------------------------
// Kernel 3: QKV projection, pure-f16 fragments.
// One wave per 16-row tile of one of {Q,K,V}: D(16x64) = xh(16x1024) @ W.
// ---------------------------------------------------------------------------
__global__ __launch_bounds__(128) void qkv_proj_kernel(
    const _Float16* __restrict__ xh,   // [BT][E] f16
    const _Float16* __restrict__ Wt,   // [3][HD][E] f16 (transposed)
    _Float16* __restrict__ Qh,         // [BT][HD] f16
    _Float16* __restrict__ Kh,         // [BT][HD] f16
    _Float16* __restrict__ Vt)         // [BB][HD][T] f16 (transposed)
{
  const int lane = threadIdx.x & 31;
  const int wav  = threadIdx.x >> 5;
  const int r    = lane & 15;    // A-fragment row / B-fragment column
  const int hh   = lane >> 4;    // half-wave selector
  const int tile = blockIdx.x * 4 + wav;   // 16-row tile, 0..BT/16-1
  const int mat  = blockIdx.y;             // 0=Q, 1=K, 2=V
  const _Float16* __restrict__ Wm = Wt + (size_t)mat * HD * E;

  const _Float16* __restrict__ xrow = xh + (size_t)(tile * 16 + r) * E;

  v8f acc[4];
  #pragma unroll
  for (int nt = 0; nt < 4; ++nt) acc[nt] = v8f{0.f,0.f,0.f,0.f,0.f,0.f,0.f,0.f};

  for (int k0 = 0; k0 < E; k0 += 32) {
    // A fragment 16x32: interleaved halves, two contiguous 16B loads.
    const v8h* a0 = (const v8h*)(xrow + k0 + hh * 8);
    const v8h* a1 = (const v8h*)(xrow + k0 + 16 + hh * 8);
    const v16h a = cat8(*a0, *a1);
    #pragma unroll
    for (int nt = 0; nt < 4; ++nt) {
      // B fragment 32x16: lane half hh covers K = hh*8..hh*8+7 and
      // 16+hh*8..16+hh*8+7 (matches A) -> two 16B loads.
      const _Float16* wp = Wm + (size_t)(nt * 16 + r) * E + k0 + hh * 8;
      v16h bw = cat8(*(const v8h*)wp, *(const v8h*)(wp + 16));
      acc[nt] = __builtin_amdgcn_wmma_f32_16x16x32_f16(
          false, a, false, bw, (short)0, acc[nt], false, false);
    }
  }

  // Store C/D fragment: lane holds col n = lane&15, rows m = v + 8*hh.
  const int n = r;
  if (mat == 2) {
    const int b = (tile * 16) / T;          // tiles never straddle a batch
    #pragma unroll
    for (int nt = 0; nt < 4; ++nt)
      #pragma unroll
      for (int v = 0; v < 8; ++v) {
        const int m = v + 8 * hh;
        const int t = tile * 16 + m - b * T;
        Vt[((size_t)b * HD + nt * 16 + n) * T + t] = (_Float16)acc[nt][v];
      }
  } else {
    _Float16* __restrict__ dst = (mat == 0) ? Qh : Kh;
    #pragma unroll
    for (int nt = 0; nt < 4; ++nt)
      #pragma unroll
      for (int v = 0; v < 8; ++v) {
        const int m = v + 8 * hh;
        dst[(size_t)(tile * 16 + m) * HD + nt * 16 + n] = (_Float16)acc[nt][v];
      }
  }
}

// ---------------------------------------------------------------------------
// Kernel 4: flash attention. 4 waves/block; each wave owns a 16-row query
// tile and streams 32-wide key chunks with online softmax + dropout.
// ---------------------------------------------------------------------------
__global__ __launch_bounds__(128) void flash_attn_kernel(
    const _Float16* __restrict__ Qh,  // [BT][HD]
    const _Float16* __restrict__ Kh,  // [BT][HD]
    const _Float16* __restrict__ Vt,  // [BB][HD][T]
    float* __restrict__ out)          // [BT][HD] f32
{
  __shared__ __align__(16) _Float16 pbuf[4][16 * 32];  // per-wave P staging

  const int lane = threadIdx.x & 31;
  const int wav  = threadIdx.x >> 5;
  const int task = blockIdx.x * 4 + wav;   // 0 .. BB*T/16 - 1
  const int b    = task >> 8;              // / (T/16)
  const int qt   = task & 255;
  const int n    = lane & 15;
  const int hh   = lane >> 4;

  const _Float16* __restrict__ Qb = Qh + (size_t)b * T * HD;
  const _Float16* __restrict__ Kb = Kh + (size_t)b * T * HD;
  const _Float16* __restrict__ Vb = Vt + (size_t)b * HD * T;

  // Q A-fragments: 2 chunks of K=32 covering HD=64. Row = lane&15.
  v16h aq[2];
  {
    const int qrow = qt * 16 + n;
    #pragma unroll
    for (int c = 0; c < 2; ++c) {
      const v8h* p0 = (const v8h*)(Qb + (size_t)qrow * HD + c * 32 + hh * 8);
      const v8h* p1 = (const v8h*)(Qb + (size_t)qrow * HD + c * 32 + 16 + hh * 8);
      aq[c] = cat8(*p0, *p1);
    }
  }

  v8f o[4];
  #pragma unroll
  for (int nt = 0; nt < 4; ++nt) o[nt] = v8f{0.f,0.f,0.f,0.f,0.f,0.f,0.f,0.f};
  float mrun[8], lrun[8];
  #pragma unroll
  for (int v = 0; v < 8; ++v) { mrun[v] = -3.0e38f; lrun[v] = 0.f; }

  const int   qbase = qt * 16;
  const int   nj    = (qbase + 16 + 31) >> 5;   // chunks up to the diagonal
  const float scale = 0.125f;                   // HD^-0.5

  for (int j = 0; j < nj; ++j) {
    const int k0 = j * 32;

    // ---- S = Q K^T : two 16x16 column tiles, each over 2 h-chunks ----
    v8f s[2];
    #pragma unroll
    for (int t = 0; t < 2; ++t) {
      v8f cf = v8f{0.f,0.f,0.f,0.f,0.f,0.f,0.f,0.f};
      const int krow = k0 + t * 16 + n;          // B-fragment column = K row
      #pragma unroll
      for (int c = 0; c < 2; ++c) {
        // B fragment 32x16: K index hh*8+i and 16+hh*8+i along the H axis.
        const _Float16* kp = Kb + (size_t)krow * HD + c * 32 + hh * 8;
        v16h bk = cat8(*(const v8h*)kp, *(const v8h*)(kp + 16));
        cf = __builtin_amdgcn_wmma_f32_16x16x32_f16(
            false, aq[c], false, bk, (short)0, cf, false, false);
      }
      s[t] = cf;
    }

    // ---- scale + causal mask ----
    const bool doMask = (k0 + 31 > qbase);
    #pragma unroll
    for (int t = 0; t < 2; ++t) {
      const int kcol = k0 + t * 16 + n;
      #pragma unroll
      for (int v = 0; v < 8; ++v) {
        float sv = s[t][v] * scale;
        if (doMask && (kcol > qbase + v + 8 * hh)) sv = -3.0e38f;
        s[t][v] = sv;
      }
    }

    // ---- online softmax (row reductions across each half-wave) ----
    float pcur[2][8];
    #pragma unroll
    for (int v = 0; v < 8; ++v) {
      float cm = fmaxf(s[0][v], s[1][v]);
      cm = fmaxf(cm, __shfl_xor(cm, 1, 32));
      cm = fmaxf(cm, __shfl_xor(cm, 2, 32));
      cm = fmaxf(cm, __shfl_xor(cm, 4, 32));
      cm = fmaxf(cm, __shfl_xor(cm, 8, 32));
      const float mnew  = fmaxf(mrun[v], cm);
      const float alpha = __expf(mrun[v] - mnew);
      const float p0 = __expf(s[0][v] - mnew);
      const float p1 = __expf(s[1][v] - mnew);
      float rs = p0 + p1;
      rs += __shfl_xor(rs, 1, 32);
      rs += __shfl_xor(rs, 2, 32);
      rs += __shfl_xor(rs, 4, 32);
      rs += __shfl_xor(rs, 8, 32);
      lrun[v] = lrun[v] * alpha + rs;   // denominator uses pre-dropout probs
      mrun[v] = mnew;
      #pragma unroll
      for (int nt = 0; nt < 4; ++nt) o[nt][v] *= alpha;
      pcur[0][v] = p0; pcur[1][v] = p1;
    }

    // ---- dropout + stage P (f16) to LDS in row-major 16x32 ----
    #pragma unroll
    for (int t = 0; t < 2; ++t) {
      const int kcol = k0 + t * 16 + n;
      #pragma unroll
      for (int v = 0; v < 8; ++v) {
        const int m    = v + 8 * hh;
        const int qrow = qbase + m;
        float pv = pcur[t][v];
        pv = keep_drop(b, qrow, kcol) ? pv * (1.0f / 0.9f) : 0.0f;
        pbuf[wav][m * 32 + t * 16 + n] = (_Float16)pv;
      }
    }
    // Wave-internal LDS RAW: DS ops are in-order per wave; compiler inserts
    // the s_wait_dscnt before the dependent reads.

    // ---- re-read P in the 16x32 A-fragment layout ----
    v16h ap;
    {
      const v8h* p0 = (const v8h*)&pbuf[wav][n * 32 + hh * 8];
      const v8h* p1 = (const v8h*)&pbuf[wav][n * 32 + 16 + hh * 8];
      ap = cat8(*p0, *p1);
    }

    // ---- O += P @ V : 4 column tiles of 16, contiguous Vt B-fragments ----
    #pragma unroll
    for (int nt = 0; nt < 4; ++nt) {
      const int hcol = nt * 16 + n;
      const _Float16* vp = Vb + (size_t)hcol * T + k0 + hh * 8;
      v16h bv = cat8(*(const v8h*)vp, *(const v8h*)(vp + 16));
      o[nt] = __builtin_amdgcn_wmma_f32_16x16x32_f16(
          false, ap, false, bv, (short)0, o[nt], false, false);
    }
  }

  // ---- normalize by row sums and store f32 output ----
  #pragma unroll
  for (int v = 0; v < 8; ++v) {
    const float inv = 1.0f / lrun[v];
    const int   m   = v + 8 * hh;
    const size_t orow = (size_t)(b * T + qbase + m);
    #pragma unroll
    for (int nt = 0; nt < 4; ++nt)
      out[orow * HD + nt * 16 + n] = o[nt][v] * inv;
  }
}

// ---------------------------------------------------------------------------
extern "C" void kernel_launch(void* const* d_in, const int* in_sizes, int n_in,
                              void* d_out, int out_size, void* d_ws, size_t ws_size,
                              hipStream_t stream) {
  const float* x  = (const float*)d_in[0];
  const float* Wq = (const float*)d_in[1];
  const float* Wk = (const float*)d_in[2];
  const float* Wv = (const float*)d_in[3];
  float* out = (float*)d_out;

  // Workspace layout (f16):
  //   Qh [BT][HD]      2 MB
  //   Kh [BT][HD]      2 MB
  //   Vt [BB][HD][T]   2 MB
  //   xh [BT][E]      32 MB
  //   Wt [3][HD][E]  384 KB
  _Float16* Qh = (_Float16*)d_ws;
  _Float16* Kh = Qh + (size_t)BT * HD;
  _Float16* Vt = Kh + (size_t)BT * HD;
  _Float16* xh = Vt + (size_t)BT * HD;
  _Float16* Wt = xh + (size_t)BT * E;

  (void)in_sizes; (void)n_in; (void)out_size; (void)ws_size;

  // Prep: convert x and W to f16 (W transposed for contiguous B-fragments).
  cvt_x_kernel<<<(BT * E / 8) / 256, 256, 0, stream>>>(x, xh);
  cvt_w_kernel<<<(3 * E * HD) / 256, 256, 0, stream>>>(Wq, Wk, Wv, Wt);

  // Projection: BT/16 = 1024 row tiles, 4 waves/block, y = {Q,K,V}.
  dim3 gProj(BT / 16 / 4, 3, 1);
  qkv_proj_kernel<<<gProj, 128, 0, stream>>>(xh, Wt, Qh, Kh, Vt);

  // Attention: BB*T/16 = 1024 query tiles, 4 waves/block.
  dim3 gAttn(BB * T / 16 / 4, 1, 1);
  flash_attn_kernel<<<gAttn, 128, 0, stream>>>(Qh, Kh, Vt, out);
}
